// RelationDistMult_12713103197296
// MI455X (gfx1250) — matrix operand order repaired and checked by reading, not compiled, not run on hardware
//
#include <hip/hip_runtime.h>

// out[n, c] = sum_h query[n,h] * M[h] * class[c,h]
//           = query @ (class * M)^T      (fold diagonal scale into the small operand)
// bf16 WMMA (v_wmma_f32_16x16x32_bf16), fp32 accumulate, double-buffered LDS for B.

typedef __attribute__((ext_vector_type(16))) __bf16 v16bf;
typedef __attribute__((ext_vector_type(8)))  __bf16 v8bf;   // 16 bytes
typedef __attribute__((ext_vector_type(8)))  float  v8f;

#define BM 128        // block rows  (N dim): 8 waves x 16-row strips
#define BN 256        // block cols  (C dim): 16 WMMA column subtiles per wave
#define BK 32         // K tile (matches WMMA K)
#define BSTRIDE 40    // bf16 elems per LDS column: 32 K + 8 pad (80B: bank-conflict-free)

// Stage one column of the B tile: 32 contiguous fp32 from class, scaled by M,
// converted to bf16, stored as 4x16B into LDS.
__device__ __forceinline__ void stage_B_col(const float* __restrict__ crow,
                                            const float* __restrict__ Md,
                                            int k0, __bf16* __restrict__ bdst)
{
    const float4* cp = (const float4*)(crow + k0);
    const float4* mp = (const float4*)(Md + k0);
    #pragma unroll
    for (int q = 0; q < 4; ++q) {          // 4 chunks of 8 elements
        float4 c0v = cp[2 * q],     c1v = cp[2 * q + 1];
        float4 m0v = mp[2 * q],     m1v = mp[2 * q + 1];
        v8bf p;
        p[0] = (__bf16)(c0v.x * m0v.x); p[1] = (__bf16)(c0v.y * m0v.y);
        p[2] = (__bf16)(c0v.z * m0v.z); p[3] = (__bf16)(c0v.w * m0v.w);
        p[4] = (__bf16)(c1v.x * m1v.x); p[5] = (__bf16)(c1v.y * m1v.y);
        p[6] = (__bf16)(c1v.z * m1v.z); p[7] = (__bf16)(c1v.w * m1v.w);
        *(v8bf*)(bdst + q * 8) = p;        // 16B DS store, 16B aligned
    }
}

__global__ __launch_bounds__(256)
void relation_distmult_wmma_bf16(const float* __restrict__ cls,   // (C, H)
                                 const float* __restrict__ qry,   // (N, H)
                                 const float* __restrict__ Md,    // (H)
                                 float* __restrict__ out,         // (N, C)
                                 int N, int C, int H)
{
    __shared__ __bf16 Bs[2 * BN * BSTRIDE];   // double buffer: 2 x 20 KB

    const int c0   = blockIdx.x * BN;
    const int m0   = blockIdx.y * BM;
    const int tid  = threadIdx.x;
    const int wave = tid >> 5;        // 0..7  -> 16-row strip per wave
    const int lane = tid & 31;
    const int r    = lane & 15;       // row within A fragment / col within B fragment
    const int hh   = lane >> 4;       // lane half selects K sub-range

    const float* qrow = qry + (size_t)(m0 + wave * 16 + r) * H;

    // B staging: each thread owns one column (128B coalesced global load per step)
    const float* crow = cls + (size_t)(c0 + tid) * H;

    __bf16* cur = Bs;
    __bf16* nxt = Bs + BN * BSTRIDE;

    v8f acc[16] = {};   // 16 column subtiles of 16x16 f32 (128 VGPRs)

    stage_B_col(crow, Md, 0, &cur[tid * BSTRIDE]);
    __syncthreads();

    for (int k0 = 0; k0 < H; k0 += BK) {
        // ---- prefetch-stage next B tile into the other buffer ----
        if (k0 + BK < H)
            stage_B_col(crow, Md, k0 + BK, &nxt[tid * BSTRIDE]);

        // ---- A fragment (16x32 bf16, §7.12.2 layout):
        //      lane half hh: K = hh*8 + {0..7}  then  hh*8 + 16 + {0..7}
        v16bf afrag;
        {
            const float4* p0 = (const float4*)(qrow + k0 + hh * 8);
            const float4* p1 = (const float4*)(qrow + k0 + hh * 8 + 16);
            float4 a0 = p0[0], a1 = p0[1];
            float4 a2 = p1[0], a3 = p1[1];
            afrag[0]  = (__bf16)a0.x; afrag[1]  = (__bf16)a0.y;
            afrag[2]  = (__bf16)a0.z; afrag[3]  = (__bf16)a0.w;
            afrag[4]  = (__bf16)a1.x; afrag[5]  = (__bf16)a1.y;
            afrag[6]  = (__bf16)a1.z; afrag[7]  = (__bf16)a1.w;
            afrag[8]  = (__bf16)a2.x; afrag[9]  = (__bf16)a2.y;
            afrag[10] = (__bf16)a2.z; afrag[11] = (__bf16)a2.w;
            afrag[12] = (__bf16)a3.x; afrag[13] = (__bf16)a3.y;
            afrag[14] = (__bf16)a3.z; afrag[15] = (__bf16)a3.w;
        }

        // ---- 16 column subtiles: B fragment (32x16): lane r = col, hh = K half ----
        #pragma unroll
        for (int j = 0; j < 16; ++j) {
            const __bf16* bsrc = &cur[(j * 16 + r) * BSTRIDE + hh * 8];
            v8bf blo = *(const v8bf*)(bsrc);
            v8bf bhi = *(const v8bf*)(bsrc + 16);
            v16bf bfrag = __builtin_shufflevector(blo, bhi,
                0, 1, 2, 3, 4, 5, 6, 7, 8, 9, 10, 11, 12, 13, 14, 15);
            acc[j] = __builtin_amdgcn_wmma_f32_16x16x32_bf16(
                false, afrag, false, bfrag, (short)0, acc[j], false, false);
        }

        __syncthreads();
        __bf16* t = cur; cur = nxt; nxt = t;
    }

    // ---- write out: C/D layout -> VGPR g: M = hh*8 + g, N = r ----
    float* obase = out + (size_t)(m0 + wave * 16 + hh * 8) * C + c0 + r;
    #pragma unroll
    for (int j = 0; j < 16; ++j) {
        #pragma unroll
        for (int g = 0; g < 8; ++g) {
            obase[(size_t)g * C + j * 16] = acc[j][g];
        }
    }
}

extern "C" void kernel_launch(void* const* d_in, const int* in_sizes, int n_in,
                              void* d_out, int out_size, void* d_ws, size_t ws_size,
                              hipStream_t stream)
{
    const float* cls = (const float*)d_in[0];   // class_vector (C, H)
    const float* qry = (const float*)d_in[1];   // query_encoder (N, H)
    const float* Md  = (const float*)d_in[2];   // M (H, 1)

    const int H = in_sizes[2];              // 2048
    const int C = in_sizes[0] / H;          // 1024
    const int N = in_sizes[1] / H;          // 16384

    dim3 grid(C / BN, N / BM);              // (4, 128)
    relation_distmult_wmma_bf16<<<grid, 256, 0, stream>>>(
        cls, qry, Md, (float*)d_out, N, C, H);
}
